// SimpleYOLOWorldDetector_63041529971152
// MI455X (gfx1250) — compile-verified, matrix operand order checked by reading, not run on hardware
//
#include <hip/hip_runtime.h>
#include <hip/hip_bf16.h>
#include <math.h>

// ---------------- static config (matches reference) ----------------
#define NPTS      8400
#define CDIM      256
#define EDIM      512
#define NCLS      80
#define NMS_PRE   1000
#define MAX_DET   300
#define SCORE_THR 0.001f
#define IOU_THR   0.65f
#define IMG       640.f
#define LSCALE    20.f
#define LBIAS     (-4.f)
#define BATCH     16

#define TILE_P    64
#define NTILES    ((NPTS + TILE_P - 1) / TILE_P)   // 132

// ---------------- workspace layout (bytes) ----------------
#define WS_WCLS   0
#define WS_TF     (WS_WCLS + EDIM * CDIM * 2)                  // 262144
#define WS_SCORES (WS_TF + NCLS * EDIM * 2)                    // 344064 (256-aligned)
#define WS_DISTS  (WS_SCORES + BATCH * NPTS * NCLS * 4)

typedef _Float16 v16h __attribute__((ext_vector_type(16)));
typedef float    v8f  __attribute__((ext_vector_type(8)));

__device__ __forceinline__ unsigned lds_off(const void* p) {
  // generic -> LDS(addrspace 3) pointer; AS3 pointers are 32-bit byte offsets
  return (unsigned)(unsigned long long)(__attribute__((address_space(3))) const char*)p;
}

__device__ __forceinline__ void async_wait0() {
#if __has_builtin(__builtin_amdgcn_s_wait_asynccnt)
  __builtin_amdgcn_s_wait_asynccnt(0);
#else
  asm volatile("s_wait_asynccnt 0" ::: "memory");
#endif
}

__device__ __forceinline__ void point_of(int n, float& cx, float& cy, float& st) {
  if (n < 6400)      { int y = n / 80;            int x = n - y * 80;          st = 8.f;  cx = (x + 0.5f) * 8.f;  cy = (y + 0.5f) * 8.f; }
  else if (n < 8000) { int m = n - 6400; int y = m / 40; int x = m - y * 40;   st = 16.f; cx = (x + 0.5f) * 16.f; cy = (y + 0.5f) * 16.f; }
  else               { int m = n - 8000; int y = m / 20; int x = m - y * 20;   st = 32.f; cx = (x + 0.5f) * 32.f; cy = (y + 0.5f) * 32.f; }
}

// ------------------------------------------------------------------
// Kernel 1: convert w_cls -> f16, normalize text_feats -> f16
// ------------------------------------------------------------------
__global__ void prep_kernel(const float* __restrict__ w_cls,
                            const float* __restrict__ text,
                            char* __restrict__ ws) {
  _Float16* wclsh = (_Float16*)(ws + WS_WCLS);
  _Float16* tfh   = (_Float16*)(ws + WS_TF);
  int tid = threadIdx.x;
  for (int i = tid; i < EDIM * CDIM; i += blockDim.x)
    wclsh[i] = (_Float16)w_cls[i];
  if (tid < NCLS) {
    float ss = 0.f;
    for (int e = 0; e < EDIM; ++e) { float v = text[tid * EDIM + e]; ss += v * v; }
    float inv = 1.f / (sqrtf(ss) + 1e-6f);
    for (int e = 0; e < EDIM; ++e)
      tfh[tid * EDIM + e] = (_Float16)(text[tid * EDIM + e] * inv);
  }
}

// ------------------------------------------------------------------
// Kernel 2: fused contrastive head + reg head for a 64-point tile.
//   feats tile staged with GLOBAL_LOAD_ASYNC_TO_LDS (ASYNCcnt path)
//   embed = feats @ w_cls^T + b_cls                      (WMMA f16, K=256)
//   scores = sigmoid((embed/||e||) @ tf_n^T * 20 - 4)    (WMMA f16, K=512)
//   dists  = softplus(feats @ w_reg^T + b_reg) * stride
// ------------------------------------------------------------------
__global__ __launch_bounds__(128, 1)
void head_kernel(const float* __restrict__ f0, const float* __restrict__ f1,
                 const float* __restrict__ f2, const float* __restrict__ b_cls,
                 const float* __restrict__ w_reg, const float* __restrict__ b_reg,
                 char* __restrict__ ws) {
  __shared__ float    sA[TILE_P][CDIM] __attribute__((aligned(32)));   // 64 KB (fp32, async-landed)
  __shared__ _Float16 sE[TILE_P][EDIM] __attribute__((aligned(32)));   // 64 KB
  __shared__ float sInv[TILE_P];

  const _Float16* wclsh = (const _Float16*)(ws + WS_WCLS);
  const _Float16* tfh   = (const _Float16*)(ws + WS_TF);
  float* scores = (float*)(ws + WS_SCORES);
  float* dists  = (float*)(ws + WS_DISTS);

  const int tile = blockIdx.x;
  const int b    = blockIdx.y;
  const int tid  = threadIdx.x;
  const int n0   = tile * TILE_P;

  // ---- stage feats tile [64 pts x 256 ch] into LDS via async copy ----
  for (int i = tid; i < TILE_P * CDIM; i += 128) {
    int p = i & (TILE_P - 1);
    int c = i >> 6;                 // TILE_P == 64
    int n = n0 + p;
    unsigned loff = lds_off(&sA[p][c]);
    if (n < NPTS) {
      const float* gp;
      if (n < 6400)      gp = f0 + ((size_t)b * CDIM + c) * 6400 + n;
      else if (n < 8000) gp = f1 + ((size_t)b * CDIM + c) * 1600 + (n - 6400);
      else               gp = f2 + ((size_t)b * CDIM + c) * 400  + (n - 8000);
      asm volatile("global_load_async_to_lds_b32 %0, %1, off"
                   :: "v"(loff), "v"(gp) : "memory");
    } else {
      sA[p][c] = 0.f;               // zero-pad tail points
    }
  }
  async_wait0();                    // drain this wave's ASYNCcnt
  __syncthreads();

  const int lane  = tid & 31;
  const int wave  = tid >> 5;          // 0..3, each owns a 16-row M-tile
  const int mrow  = lane & 15;
  const int khalf = (lane >> 4) * 16;  // lanes 16..31 hold K upper half
  const int m0    = wave * 16;

  // ---- build A fragments (fp32 LDS -> f16 regs): K=256 -> 8 steps of 32 ----
  v16h afrag[8];
#pragma unroll
  for (int k = 0; k < 8; ++k) {
    v16h a;
#pragma unroll
    for (int t = 0; t < 16; ++t)
      a[t] = (_Float16)sA[m0 + mrow][k * 32 + khalf + t];
    afrag[k] = a;
  }

  v8f ssq;
#pragma unroll
  for (int r = 0; r < 8; ++r) ssq[r] = 0.f;

  // ---- GEMM 1: embed[16 x 512] for this M-tile ----
  for (int et = 0; et < EDIM / 16; ++et) {
    v8f acc;
#pragma unroll
    for (int r = 0; r < 8; ++r) acc[r] = 0.f;
    const int e = et * 16 + mrow;                 // B column = output embed dim
    const _Float16* bp = wclsh + (size_t)e * CDIM + khalf;
    __builtin_prefetch(bp, 0, 1);                 // global_prefetch_b8
#pragma unroll
    for (int k = 0; k < 8; ++k) {
      v16h bfrag = *(const v16h*)(bp + k * 32);
      acc = __builtin_amdgcn_wmma_f32_16x16x32_f16(
          false, afrag[k], false, bfrag, (short)0, acc, false, false);
    }
    float bias = b_cls[e];
#pragma unroll
    for (int r = 0; r < 8; ++r) {
      float v = acc[r] + bias;
      ssq[r] += v * v;
      int ml = r + (lane >> 4) * 8;               // C/D layout: row = r (+8 for hi half)
      sE[m0 + ml][et * 16 + mrow] = (_Float16)v;
    }
  }

  // ---- per-row inverse L2 norm (reduce 16 lanes within each half) ----
#pragma unroll
  for (int r = 0; r < 8; ++r) {
    float s = ssq[r];
    s += __shfl_xor(s, 1, 32);
    s += __shfl_xor(s, 2, 32);
    s += __shfl_xor(s, 4, 32);
    s += __shfl_xor(s, 8, 32);
    if (mrow == 0) sInv[m0 + r + (lane >> 4) * 8] = 1.f / (sqrtf(s) + 1e-6f);
  }
  __syncthreads();

  // ---- GEMM 2: logits[16 x 80] = embed @ tf_n^T, K=512 -> 16 steps ----
  for (int ct = 0; ct < NCLS / 16; ++ct) {
    v8f acc;
#pragma unroll
    for (int r = 0; r < 8; ++r) acc[r] = 0.f;
    const _Float16* bp = tfh + (size_t)(ct * 16 + mrow) * EDIM + khalf;
    __builtin_prefetch(bp, 0, 1);
#pragma unroll
    for (int k = 0; k < EDIM / 32; ++k) {
      v16h a2 = *(const v16h*)&sE[m0 + mrow][k * 32 + khalf];
      v16h b2 = *(const v16h*)(bp + k * 32);
      acc = __builtin_amdgcn_wmma_f32_16x16x32_f16(
          false, a2, false, b2, (short)0, acc, false, false);
    }
#pragma unroll
    for (int r = 0; r < 8; ++r) {
      int ml = r + (lane >> 4) * 8;
      int p  = m0 + ml;
      int n  = n0 + p;
      if (n < NPTS) {
        float logit = acc[r] * sInv[p] * LSCALE + LBIAS;
        float sc = 1.f / (1.f + __expf(-logit));
        scores[(size_t)b * NPTS * NCLS + (size_t)n * NCLS + (ct * 16 + mrow)] = sc;
      }
    }
  }

  // ---- reg head: 2 threads per point, 2 of 4 outputs each ----
  {
    int p  = tid >> 1;
    int r0 = (tid & 1) * 2;
    int n  = n0 + p;
    if (n < NPTS) {
      float a0 = 0.f, a1 = 0.f;
      for (int c = 0; c < CDIM; ++c) {
        float f = sA[p][c];
        a0 += f * w_reg[r0 * CDIM + c];
        a1 += f * w_reg[(r0 + 1) * CDIM + c];
      }
      a0 += b_reg[r0];
      a1 += b_reg[r0 + 1];
      float cx, cy, st;
      point_of(n, cx, cy, st);
      float d0 = (a0 > 20.f) ? a0 : log1pf(__expf(a0));
      float d1 = (a1 > 20.f) ? a1 : log1pf(__expf(a1));
      dists[((size_t)b * NPTS + n) * 4 + r0]     = d0 * st;
      dists[((size_t)b * NPTS + n) * 4 + r0 + 1] = d1 * st;
    }
  }
}

// ------------------------------------------------------------------
// Kernel 3: per-image exact top-1000 (2-level bit-histogram select),
//           bitonic sort, greedy class-aware NMS, top-300 output.
// ------------------------------------------------------------------
__global__ __launch_bounds__(1024, 1)
void post_kernel(const char* __restrict__ ws, float* __restrict__ out) {
  __shared__ unsigned int h1[256];
  __shared__ unsigned int h2[2048];
  __shared__ float  sScore[1024];
  __shared__ int    sId[1024];
  __shared__ float4 sBox[NMS_PRE];
  __shared__ float  sArea[NMS_PRE];
  __shared__ short  sLab[1024];
  __shared__ unsigned char sKeep[1024];
  __shared__ int selE, selM, cntHi, cntDef, cntBd;

  const float* scores = (const float*)(ws + WS_SCORES);
  const float* dists  = (const float*)(ws + WS_DISTS);
  const int b   = blockIdx.x;
  const int tid = threadIdx.x;
  const int NK  = NPTS * NCLS;                    // 672000
  const float* sb = scores + (size_t)b * NK;

  for (int i = tid; i < MAX_DET * 6; i += 1024)
    out[(size_t)b * MAX_DET * 6 + i] = 0.f;

  if (tid < 256) h1[tid] = 0u;
  for (int i = tid; i < 2048; i += 1024) h2[i] = 0u;
  sScore[tid] = -1.f;
  sId[tid] = 0;
  if (tid == 0) { cntDef = 0; cntBd = 0; }
  __syncthreads();

  // ---- phase A: exponent histogram (positive floats are bit-monotonic) ----
  for (int i = tid; i < NK; i += 1024) {
    float s = sb[i];
    if (s > SCORE_THR) atomicAdd(&h1[__float_as_uint(s) >> 23], 1u);
  }
  __syncthreads();
  if (tid == 0) {
    int c = 0, se = 0, ch = 0;
    for (int e = 255; e >= 0; --e) {
      int hc = (int)h1[e];
      if (c + hc >= NMS_PRE || e == 0) { se = e; ch = c; break; }
      c += hc;
    }
    selE = se; cntHi = ch;
  }
  __syncthreads();
  const int sEsel = selE;

  // ---- phase B: refine inside the boundary exponent, 11 mantissa bits ----
  for (int i = tid; i < NK; i += 1024) {
    float s = sb[i];
    if (s > SCORE_THR) {
      unsigned bits = __float_as_uint(s);
      if ((int)(bits >> 23) == sEsel)
        atomicAdd(&h2[(bits >> 12) & 0x7FF], 1u);
    }
  }
  __syncthreads();
  if (tid == 0) {
    int c = cntHi, sm = 0;
    for (int m = 2047; m >= 0; --m) {
      int hc = (int)h2[m];
      if (c + hc >= NMS_PRE || m == 0) { sm = m; break; }
      c += hc;
    }
    selM = sm;
  }
  __syncthreads();
  const int sMsel = selM;

  // ---- phase C1: gather strictly-above-cut candidates (< 1000 guaranteed) ----
  for (int i = tid; i < NK; i += 1024) {
    float s = sb[i];
    if (s > SCORE_THR) {
      unsigned bits = __float_as_uint(s);
      int e = (int)(bits >> 23);
      int m = (int)((bits >> 12) & 0x7FF);
      if (e > sEsel || (e == sEsel && m > sMsel)) {
        int slot = atomicAdd(&cntDef, 1);
        if (slot < NMS_PRE) { sScore[slot] = s; sId[slot] = i; }
      }
    }
  }
  __syncthreads();
  const int base = (cntDef < NMS_PRE) ? cntDef : NMS_PRE;
  const int rem  = NMS_PRE - base;
  __syncthreads();

  // ---- phase C2: fill remaining slots from the boundary sub-bin ----
  for (int i = tid; i < NK; i += 1024) {
    float s = sb[i];
    if (s > SCORE_THR) {
      unsigned bits = __float_as_uint(s);
      int e = (int)(bits >> 23);
      int m = (int)((bits >> 12) & 0x7FF);
      if (e == sEsel && m == sMsel) {
        int slot = atomicAdd(&cntBd, 1);
        if (slot < rem) { sScore[base + slot] = s; sId[base + slot] = i; }
      }
    }
  }
  __syncthreads();

  // ---- bitonic sort 1024 entries: score desc, index asc (deterministic) ----
  for (int k = 2; k <= 1024; k <<= 1) {
    for (int j = k >> 1; j > 0; j >>= 1) {
      int ixj = tid ^ j;
      if (ixj > tid) {
        float s1 = sScore[tid], s2 = sScore[ixj];
        int   i1 = sId[tid],    i2 = sId[ixj];
        bool first_better = (s1 > s2) || (s1 == s2 && i1 < i2);
        bool desc_region  = ((tid & k) == 0);
        if (desc_region ? !first_better : first_better) {
          sScore[tid] = s2; sScore[ixj] = s1;
          sId[tid] = i2;    sId[ixj] = i1;
        }
      }
      __syncthreads();
    }
  }

  // ---- decode boxes for the top candidates ----
  if (tid < NMS_PRE) {
    float s = sScore[tid];
    bool v = s > SCORE_THR;
    sKeep[tid] = v ? 1 : 0;
    float4 bx = make_float4(0.f, 0.f, 0.f, 0.f);
    short lab = 0;
    float area = 0.f;
    if (v) {
      int id = sId[tid];
      int n  = id / NCLS;
      lab = (short)(id - n * NCLS);
      float cx, cy, st;
      point_of(n, cx, cy, st);
      const float* dp = dists + ((size_t)b * NPTS + n) * 4;
      float x1 = fminf(fmaxf(cx - dp[0], 0.f), IMG);
      float y1 = fminf(fmaxf(cy - dp[1], 0.f), IMG);
      float x2 = fminf(fmaxf(cx + dp[2], 0.f), IMG);
      float y2 = fminf(fmaxf(cy + dp[3], 0.f), IMG);
      bx = make_float4(x1, y1, x2, y2);
      area = fmaxf(x2 - x1, 0.f) * fmaxf(y2 - y1, 0.f);
    }
    sBox[tid] = bx; sLab[tid] = lab; sArea[tid] = area;
  } else {
    sKeep[tid] = 0; sLab[tid] = 0;
  }
  __syncthreads();

  // ---- greedy class-aware NMS (class offset == labels must match) ----
  for (int i = 0; i < NMS_PRE; ++i) {
    __syncthreads();
    if (!sKeep[i]) continue;                      // uniform branch
    int j = tid;
    if (j > i && j < NMS_PRE && sKeep[j] && sLab[j] == sLab[i]) {
      float4 A = sBox[i], Bb = sBox[j];
      float lx = fmaxf(A.x, Bb.x), ly = fmaxf(A.y, Bb.y);
      float rx = fminf(A.z, Bb.z), ry = fminf(A.w, Bb.w);
      float iw = fmaxf(rx - lx, 0.f), ih = fmaxf(ry - ly, 0.f);
      float inter = iw * ih;
      float iou = inter / (sArea[i] + sArea[j] - inter + 1e-7f);
      if (iou > IOU_THR) sKeep[j] = 0;
    }
  }
  __syncthreads();

  // ---- compact kept detections (already score-sorted) into top-300 ----
  if (tid == 0) {
    float* ob = out + (size_t)b * MAX_DET * 6;
    int outn = 0;
    for (int t = 0; t < NMS_PRE && outn < MAX_DET; ++t) {
      if (sKeep[t]) {
        ob[outn * 6 + 0] = sBox[t].x;
        ob[outn * 6 + 1] = sBox[t].y;
        ob[outn * 6 + 2] = sBox[t].z;
        ob[outn * 6 + 3] = sBox[t].w;
        ob[outn * 6 + 4] = sScore[t];
        ob[outn * 6 + 5] = (float)sLab[t];
        ++outn;
      }
    }
  }
}

// ------------------------------------------------------------------
extern "C" void kernel_launch(void* const* d_in, const int* in_sizes, int n_in,
                              void* d_out, int out_size, void* d_ws, size_t ws_size,
                              hipStream_t stream) {
  const float* f0     = (const float*)d_in[0];
  const float* f1     = (const float*)d_in[1];
  const float* f2     = (const float*)d_in[2];
  const float* w_cls  = (const float*)d_in[3];
  const float* b_cls  = (const float*)d_in[4];
  const float* w_reg  = (const float*)d_in[5];
  const float* b_reg  = (const float*)d_in[6];
  const float* text   = (const float*)d_in[7];
  char*  ws  = (char*)d_ws;
  float* out = (float*)d_out;

  prep_kernel<<<dim3(1), dim3(256), 0, stream>>>(w_cls, text, ws);
  head_kernel<<<dim3(NTILES, BATCH), dim3(128), 0, stream>>>(
      f0, f1, f2, b_cls, w_reg, b_reg, ws);
  post_kernel<<<dim3(BATCH), dim3(1024), 0, stream>>>(ws, out);
}
